// AttnBlockpp_80831284510777
// MI455X (gfx1250) — matrix operand order repaired and checked by reading, not compiled, not run on hardware
//
#include <hip/hip_runtime.h>

typedef __attribute__((ext_vector_type(16))) _Float16 v16h;
typedef __attribute__((ext_vector_type(8)))  _Float16 v8h;
typedef __attribute__((ext_vector_type(8)))  float    v8f;

#define HW   4096
#define CDIM 256
#define BATCH 4

// ---------------- WMMA fragment helpers (CDNA5 16x16x32 f16) ----------------
// A (16x32 f16): lane m = lane&15, hi = lane>>4.
//   elems 0..7  -> K = hi*8 + e          (contiguous run)
//   elems 8..15 -> K = 16 + hi*8 + (e-8) (contiguous run)
__device__ __forceinline__ v16h load_a_frag(const _Float16* p, int ld, int row, int col) {
    const int lane = threadIdx.x & 31;
    const int m = lane & 15, hi = lane >> 4;
    const _Float16* r = p + (size_t)(row + m) * ld + col + hi * 8;
    v8h lo = *(const v8h*)r;
    v8h hh = *(const v8h*)(r + 16);
    v16h out;
#pragma unroll
    for (int i = 0; i < 8; ++i) { out[i] = lo[i]; out[i + 8] = hh[i]; }
    return out;
}

// B (32x16 f16): lane n = lane&15, elems 0..7 -> K = hi*8 + e, elems 8..15 -> K = 16 + hi*8 + (e-8).
// Source buffer laid out [n][k] row-major with leading dim ld.
__device__ __forceinline__ v16h load_b_frag(const _Float16* p, int ld, int n0, int k0) {
    const int lane = threadIdx.x & 31;
    const int n = lane & 15, hi = lane >> 4;
    const _Float16* r = p + (size_t)(n0 + n) * ld + k0 + hi * 8;
    v8h lo = *(const v8h*)r;
    v8h hh = *(const v8h*)(r + 16);
    v16h out;
#pragma unroll
    for (int i = 0; i < 8; ++i) { out[i] = lo[i]; out[i + 8] = hh[i]; }
    return out;
}

__device__ __forceinline__ v8f wmma_f16(v16h a, v16h b, v8f c) {
    return __builtin_amdgcn_wmma_f32_16x16x32_f16(false, a, false, b, (short)0, c, false, false);
}

// Generic pointer to LDS: low 32 bits of the flat address ARE the LDS byte
// offset (ISA 10.2: LDS aperture lives in addr[63:32]).
__device__ __forceinline__ unsigned lds_off(const void* p) {
    return (unsigned)(unsigned long long)p;
}

// Async copy 16B/lane global -> LDS (ASYNCcnt-tracked, gfx1250).
__device__ __forceinline__ void async_ld_b128(unsigned lds, const void* gptr) {
    asm volatile("global_load_async_to_lds_b128 %0, %1, off"
                 :: "v"(lds), "v"(gptr) : "memory");
}

// ---------------- Kernel 1: weight transpose + f16 convert ----------------
__global__ __launch_bounds__(256) void wt_kernel(const float* __restrict__ w0,
                                                 const float* __restrict__ w1,
                                                 const float* __restrict__ w2,
                                                 const float* __restrict__ w3,
                                                 _Float16* __restrict__ wt) {
    const int mat = blockIdx.x >> 8;
    const int d   = blockIdx.x & 255;
    const int c   = threadIdx.x;
    const float* w = (mat == 0) ? w0 : (mat == 1) ? w1 : (mat == 2) ? w2 : w3;
    wt[((size_t)mat * CDIM + d) * CDIM + c] = (_Float16)w[(size_t)c * CDIM + d];
}

// ---------------- Kernel 2: GroupNorm -> h (f16, [B][HW][C]) ----------------
__global__ __launch_bounds__(256) void gn_kernel(const float* __restrict__ x,
                                                 const float* __restrict__ gamma,
                                                 const float* __restrict__ beta,
                                                 _Float16* __restrict__ ht) {
    const int b = blockIdx.x >> 5;
    const int g = blockIdx.x & 31;
    const size_t base = ((size_t)b * CDIM + g * 8) * HW;
    const int N = 8 * HW;
    float s = 0.f, ss = 0.f;
    for (int i = threadIdx.x; i < N; i += 256) {
        float v = x[base + i];
        s += v; ss += v * v;
    }
    __shared__ float red[2][256];
    red[0][threadIdx.x] = s; red[1][threadIdx.x] = ss;
    __syncthreads();
    for (int off = 128; off > 0; off >>= 1) {
        if (threadIdx.x < off) {
            red[0][threadIdx.x] += red[0][threadIdx.x + off];
            red[1][threadIdx.x] += red[1][threadIdx.x + off];
        }
        __syncthreads();
    }
    const float mean = red[0][0] * (1.f / N);
    const float var  = red[1][0] * (1.f / N) - mean * mean;
    const float rstd = rsqrtf(var + 1e-6f);
    for (int i = threadIdx.x; i < N; i += 256) {
        const int ch = i >> 12, hw = i & (HW - 1);
        const int c = g * 8 + ch;
        const float v = (x[base + i] - mean) * rstd * gamma[c] + beta[c];
        ht[((size_t)b * HW + hw) * CDIM + c] = (_Float16)v;
    }
}

// ---------------- Kernel 3: fused Q/K/V projection (WMMA GEMM) ----------------
__global__ __launch_bounds__(32) void qkv_kernel(const _Float16* __restrict__ ht,
                                                 const _Float16* __restrict__ wt,
                                                 const float* __restrict__ b0,
                                                 const float* __restrict__ b1,
                                                 const float* __restrict__ b2,
                                                 _Float16* __restrict__ qo,
                                                 _Float16* __restrict__ ko,
                                                 _Float16* __restrict__ vo) {
    const int hw0 = blockIdx.x * 64;
    const int d0  = blockIdx.y * 16;
    const int b   = blockIdx.z / 3;
    const int mat = blockIdx.z % 3;
    const _Float16* hb = ht + (size_t)b * HW * CDIM;
    const _Float16* wm = wt + (size_t)mat * CDIM * CDIM;
    const float* bias = (mat == 0) ? b0 : (mat == 1) ? b1 : b2;

    v8f acc[4] = {};
#pragma unroll
    for (int kc = 0; kc < 8; ++kc) {
        v16h a = load_a_frag(wm, CDIM, d0, kc * 32);
#pragma unroll
        for (int t = 0; t < 4; ++t) {
            v16h bm = load_b_frag(hb, CDIM, hw0 + t * 16, kc * 32);
            acc[t] = wmma_f16(a, bm, acc[t]);
        }
    }
    const int lane = threadIdx.x & 31;
    const int nn = lane & 15, hi = lane >> 4;
    float bv[8];
#pragma unroll
    for (int r = 0; r < 8; ++r) bv[r] = bias[d0 + hi * 8 + r];

    if (mat < 2) {
        _Float16* dst = ((mat == 0) ? qo : ko) + (size_t)b * HW * CDIM;
#pragma unroll
        for (int t = 0; t < 4; ++t) {
            const int hw = hw0 + t * 16 + nn;
            v8h pk;
#pragma unroll
            for (int r = 0; r < 8; ++r) pk[r] = (_Float16)(acc[t][r] + bv[r]);
            *(v8h*)(dst + (size_t)hw * CDIM + d0 + hi * 8) = pk;
        }
    } else {
        _Float16* dst = vo + (size_t)b * CDIM * HW;
#pragma unroll
        for (int t = 0; t < 4; ++t) {
            const int hw = hw0 + t * 16 + nn;
#pragma unroll
            for (int r = 0; r < 8; ++r)
                dst[(size_t)(d0 + hi * 8 + r) * HW + hw] = (_Float16)(acc[t][r] + bv[r]);
        }
    }
}

// ---------------- Kernel 4: flash attention ----------------
// 4-wave workgroup: each wave owns 16 queries; K/V tiles (32 keys) are shared
// through LDS, filled by async global->LDS DMA, double-buffered so tile j+1
// streams in while tile j is consumed. ASYNCcnt + workgroup barrier sync.
//
// LDS layout (halves), single flat buffer:
//   Kt(buf,key,c) = buf*8448  + key*264 + c        rows padded to 528B
//   Vt(buf,c,k)   = 16896 + buf*10240 + c*40 + k   rows padded to 80B
//   Pl(w,q,k)     = 37376 + w*640 + q*40 + k       per-wave P staging
#define SM_KT(buf, key, c) ((buf) * 8448 + (key) * 264 + (c))
#define SM_VT(buf, c, k)   (16896 + (buf) * 10240 + (c) * 40 + (k))
#define SM_PL(w, q, k)     (37376 + (w) * 640 + (q) * 40 + (k))
#define SMEM_HALVES        39936

__global__ __launch_bounds__(128) void attn_kernel(const _Float16* __restrict__ qt,
                                                   const _Float16* __restrict__ kt,
                                                   const _Float16* __restrict__ vm,
                                                   _Float16* __restrict__ ho) {
    __shared__ __align__(16) _Float16 smem[SMEM_HALVES];

    const int b    = blockIdx.y;
    const int wave = threadIdx.x >> 5;
    const int tid  = threadIdx.x;
    const int lane = threadIdx.x & 31;
    const int nn = lane & 15, hi = lane >> 4;
    const int q0 = blockIdx.x * 64 + wave * 16;

    const _Float16* qb = qt + (size_t)b * HW * CDIM;
    const _Float16* kb = kt + (size_t)b * HW * CDIM;
    const _Float16* vb = vm + (size_t)b * CDIM * HW;

    // ---- async tile fetch: 32 keys of K ([key][C]) and V ([C][key]) ----
    auto issue_tile = [&](int j, int buf) {
#pragma unroll
        for (int it = 0; it < 8; ++it) {            // K: 32*512B = 1024 x16B chunks
            const int id  = it * 128 + tid;
            const int key = id >> 5, c16 = id & 31;
            async_ld_b128(lds_off(&smem[SM_KT(buf, key, c16 * 8)]),
                          kb + (size_t)(j + key) * CDIM + c16 * 8);
        }
#pragma unroll
        for (int it = 0; it < 8; ++it) {            // V: 256*64B = 1024 x16B chunks
            const int id = it * 128 + tid;
            const int c = id >> 2, k8 = id & 3;
            async_ld_b128(lds_off(&smem[SM_VT(buf, c, k8 * 8)]),
                          vb + (size_t)c * HW + j + k8 * 8);
        }
    };

    v16h qa[8];
#pragma unroll
    for (int kc = 0; kc < 8; ++kc) qa[kc] = load_a_frag(qb, CDIM, q0, kc * 32);

    v8f o[16];
#pragma unroll
    for (int t = 0; t < 16; ++t) { v8f z = {}; o[t] = z; }
    float m[8], l[8];
#pragma unroll
    for (int r = 0; r < 8; ++r) { m[r] = -1e30f; l[r] = 0.f; }

    const int NT = HW / 32;            // 128 key tiles
    issue_tile(0, 0);                  // prologue: two tiles in flight
    issue_tile(32, 1);

    for (int jt = 0; jt < NT; ++jt) {
        const int buf = jt & 1;
        // wait for tile jt (16 async ops/wave/tile; completion is in-order)
        if (jt + 1 < NT) asm volatile("s_wait_asynccnt 0x10" ::: "memory");
        else             asm volatile("s_wait_asynccnt 0x0"  ::: "memory");
        __syncthreads();               // all waves' chunks of tile jt visible

        // ---- scores S[16q x 32k] from LDS K tile ----
        v8f s0 = {}, s1 = {};
#pragma unroll
        for (int kc = 0; kc < 8; ++kc) {
            v16h bk0 = load_b_frag(&smem[SM_KT(buf, 0, 0)], 264, 0,  kc * 32);
            v16h bk1 = load_b_frag(&smem[SM_KT(buf, 0, 0)], 264, 16, kc * 32);
            s0 = wmma_f16(qa[kc], bk0, s0);
            s1 = wmma_f16(qa[kc], bk1, s1);
        }
        const float scale = 0.0625f;   // 256^-0.5
        float p0[8], p1[8], mt[8];
#pragma unroll
        for (int r = 0; r < 8; ++r) {
            p0[r] = s0[r] * scale; p1[r] = s1[r] * scale;
            mt[r] = fmaxf(p0[r], p1[r]);
        }
#pragma unroll
        for (int w = 1; w <= 8; w <<= 1) {
#pragma unroll
            for (int r = 0; r < 8; ++r) mt[r] = fmaxf(mt[r], __shfl_xor(mt[r], w, 32));
        }
        float alpha[8], rs[8];
#pragma unroll
        for (int r = 0; r < 8; ++r) {
            const float mn = fmaxf(m[r], mt[r]);
            alpha[r] = __expf(m[r] - mn);
            m[r] = mn;
            p0[r] = __expf(p0[r] - mn);
            p1[r] = __expf(p1[r] - mn);
            rs[r] = p0[r] + p1[r];
        }
#pragma unroll
        for (int w = 1; w <= 8; w <<= 1) {
#pragma unroll
            for (int r = 0; r < 8; ++r) rs[r] += __shfl_xor(rs[r], w, 32);
        }
#pragma unroll
        for (int r = 0; r < 8; ++r) l[r] = l[r] * alpha[r] + rs[r];
#pragma unroll
        for (int t = 0; t < 16; ++t)
#pragma unroll
            for (int r = 0; r < 8; ++r) o[t][r] *= alpha[r];

        // P: D-layout -> per-wave LDS staging -> A-layout fragment
#pragma unroll
        for (int r = 0; r < 8; ++r) {
            smem[SM_PL(wave, r + 8 * hi, nn)]      = (_Float16)p0[r];
            smem[SM_PL(wave, r + 8 * hi, nn + 16)] = (_Float16)p1[r];
        }
        // intra-wave LDS RAW: compiler inserts s_wait_dscnt (cross-lane within wave)
        v16h pa = load_a_frag(&smem[SM_PL(wave, 0, 0)], 40, 0, 0);

        // ---- O[16q x 256c] += P @ V from LDS V tile ----
#pragma unroll
        for (int t = 0; t < 16; ++t) {
            v16h vf = load_b_frag(&smem[SM_VT(buf, 0, 0)], 40, t * 16, 0);
            o[t] = wmma_f16(pa, vf, o[t]);
        }

        __syncthreads();               // everyone done with buf
        if (jt + 2 < NT) issue_tile((jt + 2) * 32, buf);  // refill freed buffer
    }

    float inv[8];
#pragma unroll
    for (int r = 0; r < 8; ++r) inv[r] = 1.f / l[r];
    _Float16* hb = ho + (size_t)b * HW * CDIM;
#pragma unroll
    for (int t = 0; t < 16; ++t)
#pragma unroll
        for (int r = 0; r < 8; ++r)
            hb[(size_t)(q0 + r + 8 * hi) * CDIM + t * 16 + nn] =
                (_Float16)(o[t][r] * inv[r]);
}

// ---------------- Kernel 5: output projection + bias + residual ----------------
__global__ __launch_bounds__(32) void out_kernel(const _Float16* __restrict__ ha,
                                                 const _Float16* __restrict__ wt3,
                                                 const float* __restrict__ b3,
                                                 const float* __restrict__ x,
                                                 float* __restrict__ out) {
    const int hw0 = blockIdx.x * 64;
    const int d0  = blockIdx.y * 16;
    const int b   = blockIdx.z;
    const _Float16* hb = ha + (size_t)b * HW * CDIM;

    v8f acc[4] = {};
#pragma unroll
    for (int kc = 0; kc < 8; ++kc) {
        v16h a = load_a_frag(wt3, CDIM, d0, kc * 32);
#pragma unroll
        for (int t = 0; t < 4; ++t) {
            v16h bm = load_b_frag(hb, CDIM, hw0 + t * 16, kc * 32);
            acc[t] = wmma_f16(a, bm, acc[t]);
        }
    }
    const int lane = threadIdx.x & 31;
    const int nn = lane & 15, hi = lane >> 4;
#pragma unroll
    for (int t = 0; t < 4; ++t) {
        const int hw = hw0 + t * 16 + nn;
#pragma unroll
        for (int r = 0; r < 8; ++r) {
            const size_t off = ((size_t)b * CDIM + d0 + hi * 8 + r) * HW + hw;
            out[off] = x[off] + b3[d0 + hi * 8 + r] + acc[t][r];
        }
    }
}

// ---------------- launcher ----------------
extern "C" void kernel_launch(void* const* d_in, const int* in_sizes, int n_in,
                              void* d_out, int out_size, void* d_ws, size_t ws_size,
                              hipStream_t stream) {
    (void)in_sizes; (void)n_in; (void)out_size; (void)ws_size;
    const float* x     = (const float*)d_in[0];
    const float* gamma = (const float*)d_in[1];
    const float* beta  = (const float*)d_in[2];
    const float* w0 = (const float*)d_in[3];
    const float* b0 = (const float*)d_in[4];
    const float* w1 = (const float*)d_in[5];
    const float* b1 = (const float*)d_in[6];
    const float* w2 = (const float*)d_in[7];
    const float* b2 = (const float*)d_in[8];
    const float* w3 = (const float*)d_in[9];
    const float* b3 = (const float*)d_in[10];
    float* out = (float*)d_out;

    const size_t nBHWC = (size_t)BATCH * HW * CDIM;
    _Float16* ws  = (_Float16*)d_ws;
    _Float16* ht  = ws;
    _Float16* qt  = ht + nBHWC;
    _Float16* kt  = qt + nBHWC;
    _Float16* vv  = kt + nBHWC;
    _Float16* wts = vv + nBHWC;
    _Float16* ha  = ht;                // h consumed before attention writes

    wt_kernel <<<dim3(4 * 256), 256, 0, stream>>>(w0, w1, w2, w3, wts);
    gn_kernel <<<dim3(BATCH * 32), 256, 0, stream>>>(x, gamma, beta, ht);
    qkv_kernel<<<dim3(HW / 64, CDIM / 16, 3 * BATCH), 32, 0, stream>>>(
        ht, wts, b0, b1, b2, qt, kt, vv);
    attn_kernel<<<dim3(HW / 64, BATCH), 128, 0, stream>>>(qt, kt, vv, ha);
    out_kernel<<<dim3(HW / 64, CDIM / 16, BATCH), 32, 0, stream>>>(
        ha, wts + (size_t)3 * CDIM * CDIM, b3, x, out);
}
